// PlainSelfAttention_47691316855182
// MI455X (gfx1250) — matrix operand order repaired and checked by reading, not compiled, not run on hardware
//
#include <hip/hip_runtime.h>

typedef __attribute__((ext_vector_type(16))) __bf16 v16bf;
typedef __attribute__((ext_vector_type(8)))  __bf16 v8bf;
typedef __attribute__((ext_vector_type(4)))  __bf16 v4bf;
typedef __attribute__((ext_vector_type(8)))  float  v8f;
typedef __attribute__((ext_vector_type(4)))  float  v4f;
typedef __attribute__((ext_vector_type(4)))  unsigned int v4u;
typedef __attribute__((ext_vector_type(8)))  int    v8i;
typedef __attribute__((ext_vector_type(4)))  int    v4i;

#define WMMA_BF16(A, B, C)                                                   \
  __builtin_amdgcn_wmma_f32_16x16x32_bf16(false, (A), false, (B), (short)0,  \
                                          (C), false, false)

static __device__ __forceinline__ v16bf make_v16(v8bf lo, v8bf hi) {
  v16bf r;
#pragma unroll
  for (int j = 0; j < 8; ++j) { r[j] = lo[j]; r[j + 8] = hi[j]; }
  return r;
}

// ---------------------------------------------------------------------------
// Tensor Data Mover: 2-D tile load (global -> LDS) with LDS row padding.
//   lds_off          : byte offset of destination inside the group segment
//   gaddr            : global address of tile start
//   tile_w / tile_h  : tile dims in elements (dim0 = contiguous)
//   stride_elems     : tensor dim0 stride in elements
//   pad_int_code     : DWORDs between pads: 0=2,1=4,2=8,3=16,4=32,...
//   pad_amt_code     : pad DWORDs: 0=1,1=2,2=3,3=4,...
// data_size = 2 bytes (bf16). Groups 2/3 zero (2-D tensor, tile_dim2=0).
// This toolchain exposes the 6-arg (clang-23) builtin form.
// ---------------------------------------------------------------------------
static __device__ __forceinline__ void tdm_load_2d(unsigned lds_off,
                                                   const void* gaddr,
                                                   unsigned tile_w,
                                                   unsigned tile_h,
                                                   unsigned stride_elems,
                                                   unsigned pad_int_code,
                                                   unsigned pad_amt_code) {
  unsigned long long ga = (unsigned long long)gaddr;
  v4u g0;
  g0[0] = 1u;                                   // count=1, user descriptor
  g0[1] = lds_off;                              // lds_addr
  g0[2] = (unsigned)(ga & 0xffffffffu);         // global_addr[31:0]
  g0[3] = (unsigned)(ga >> 32) | 0x80000000u;   // global_addr[56:32] | type=2
  v8i g1;
  g1[0] = (int)((1u << 16) |                    // data_size = 2 bytes
                (1u << 20) |                    // pad_enable
                (pad_int_code << 22) | (pad_amt_code << 25));
  g1[1] = (int)((tile_w & 0xffffu) << 16);                       // tensor_dim0 lo
  g1[2] = (int)(((tile_w >> 16) & 0xffffu) |
                ((tile_h & 0xffffu) << 16));                     // dim0 hi | dim1 lo
  g1[3] = (int)(((tile_h >> 16) & 0xffffu) | (tile_w << 16));    // dim1 hi | tile_dim0
  g1[4] = (int)(tile_h & 0xffffu);                               // tile_dim1, tile_dim2=0
  g1[5] = (int)stride_elems;                                     // dim0_stride lo32
  g1[6] = 0;
  g1[7] = 0;
  v4i z4 = {0, 0, 0, 0};
  v8i z8 = {0, 0, 0, 0, 0, 0, 0, 0};
  __builtin_amdgcn_tensor_load_to_lds(g0, g1, z4, z4, z8, 0);
}

// ---------------------------------------------------------------------------
// f32 -> bf16 conversion (4 elements / thread)
// ---------------------------------------------------------------------------
__global__ void cvt_f32_to_bf16(const float* __restrict__ s,
                                __bf16* __restrict__ d, int n) {
  int i = (blockIdx.x * blockDim.x + threadIdx.x) * 4;
  if (i >= n) return;
  v4f x = *(const v4f*)(s + i);
  v4bf y;
#pragma unroll
  for (int j = 0; j < 4; ++j) y[j] = (__bf16)x[j];
  *(v4bf*)(d + i) = y;
}

// ---------------------------------------------------------------------------
// Tiled WMMA GEMM: C[M,N] = A[M,K] * B[K,N], A/B bf16 row-major.
// Block tile 128x64, BK=32, 256 threads = 8 waves (4x2), 32x32 slab per wave
// (4 accumulators, each A/B fragment reused twice).
// A tile streamed by the Tensor Data Mover (double-buffered, s_wait_tensorcnt);
// B tile double-buffered through registers (needs a transpose TDM can't do).
// ---------------------------------------------------------------------------
template <bool OUT_BF16>
__global__ __launch_bounds__(256) void gemm_bf16_kernel(
    const __bf16* __restrict__ A, const __bf16* __restrict__ B,
    void* __restrict__ Cv, int M, int N, int K) {
  (void)M;
  struct Smem {
    __bf16 As[2][128 * 40];  // [buf][row][32 k + 8 pad]  (TDM fills, padded)
    __bf16 Bt[2][64 * 40];   // [buf][n][32 k + 8 pad]    (transposed B tile)
  };
  __shared__ Smem sm;
  const unsigned kAsOff[2] = {0u, 128u * 40u * 2u};  // byte offsets in LDS

  const int tid = threadIdx.x;
  const int wave = tid >> 5, lane = tid & 31;
  const int lh = lane >> 4, l16 = lane & 15;
  const int bm = blockIdx.y * 128, bn = blockIdx.x * 64;
  const int wm = (wave & 3) * 32, wn = (wave >> 2) * 32;

  const int nk = K >> 5;
  const int br = tid >> 3, bc = (tid & 7) * 8;  // B-tile load coords

  // prologue: TDM A[0]; cooperative B[0] -> LDS buf 0
  if (tid == 0)
    tdm_load_2d(kAsOff[0], A + (size_t)bm * K, 32, 128, K, /*16DW*/ 3, /*4DW*/ 3);
  {
    v8bf bb = *(const v8bf*)(B + (size_t)br * N + bn + bc);
#pragma unroll
    for (int j = 0; j < 8; ++j) sm.Bt[0][(bc + j) * 40 + br] = bb[j];
  }

  v8f c00 = {}, c01 = {}, c10 = {}, c11 = {};
  for (int ki = 0; ki < nk; ++ki) {
    const int buf = ki & 1;
    v8bf bb = {};
    if (ki + 1 < nk) {  // prefetch next B tile into registers
      bb = *(const v8bf*)(B + (size_t)((ki + 1) * 32 + br) * N + bn + bc);
    }
    if (tid == 0) {
      if (ki + 1 < nk) {  // TDM next A tile, then wait for current one
        tdm_load_2d(kAsOff[1 - buf], A + (size_t)bm * K + (ki + 1) * 32, 32,
                    128, K, 3, 3);
        __builtin_amdgcn_s_wait_tensorcnt((short)1);
      } else {
        __builtin_amdgcn_s_wait_tensorcnt((short)0);
      }
    }
    __syncthreads();  // A[buf] landed (wave0 waited); B[buf] staged last iter

    const __bf16* Asb = sm.As[buf];
    const __bf16* Btb = sm.Bt[buf];
    const __bf16* ap0 = &Asb[(wm + l16) * 40 + lh * 8];
    v16bf af0 = make_v16(*(const v8bf*)ap0, *(const v8bf*)(ap0 + 16));
    const __bf16* ap1 = &Asb[(wm + 16 + l16) * 40 + lh * 8];
    v16bf af1 = make_v16(*(const v8bf*)ap1, *(const v8bf*)(ap1 + 16));
    const __bf16* bp0 = &Btb[(wn + l16) * 40 + lh * 8];
    v16bf bf0 = make_v16(*(const v8bf*)bp0, *(const v8bf*)(bp0 + 16));
    const __bf16* bp1 = &Btb[(wn + 16 + l16) * 40 + lh * 8];
    v16bf bf1 = make_v16(*(const v8bf*)bp1, *(const v8bf*)(bp1 + 16));

    c00 = WMMA_BF16(af0, bf0, c00);
    c01 = WMMA_BF16(af0, bf1, c01);
    c10 = WMMA_BF16(af1, bf0, c10);
    c11 = WMMA_BF16(af1, bf1, c11);

    if (ki + 1 < nk) {  // stage next B tile into the other buffer
#pragma unroll
      for (int j = 0; j < 8; ++j) sm.Bt[1 - buf][(bc + j) * 40 + br] = bb[j];
    }
    __syncthreads();  // B staged + compute done before buffers are reused
  }

  const int rowA = bm + wm + lh * 8;
  const int col0 = bn + wn + l16;
  if (OUT_BF16) {
    __bf16* C = (__bf16*)Cv;
#pragma unroll
    for (int i = 0; i < 8; ++i) {
      C[(size_t)(rowA + i) * N + col0] = (__bf16)c00[i];
      C[(size_t)(rowA + i) * N + col0 + 16] = (__bf16)c01[i];
      C[(size_t)(rowA + 16 + i) * N + col0] = (__bf16)c10[i];
      C[(size_t)(rowA + 16 + i) * N + col0 + 16] = (__bf16)c11[i];
    }
  } else {
    float* C = (float*)Cv;
#pragma unroll
    for (int i = 0; i < 8; ++i) {
      C[(size_t)(rowA + i) * N + col0] = c00[i];
      C[(size_t)(rowA + i) * N + col0 + 16] = c01[i];
      C[(size_t)(rowA + 16 + i) * N + col0] = c10[i];
      C[(size_t)(rowA + 16 + i) * N + col0 + 16] = c11[i];
    }
  }
}

// ---------------------------------------------------------------------------
// 2Quad attention. grid = B*H*(T/128) = 512 blocks, 256 threads (8 waves).
// Each wave owns a 16-row query tile. Keys streamed in 64-key chunks by the
// Tensor Data Mover (padded rows), V transposed cooperatively. Two passes:
//   pass1: row sums of (s/8 + 5)^2 via WMMA + shfl_xor reduce (wave32 halves)
//   pass2: recompute S, write normalized prob (f32), stage bf16 prob tile in
//          LDS to form A fragments, accumulate prob @ V via WMMA.
// ---------------------------------------------------------------------------
__global__ __launch_bounds__(256) void attn2quad_kernel(
    const __bf16* __restrict__ qkv, float* __restrict__ prob,
    __bf16* __restrict__ attn) {
  struct Smem {
    __bf16 Kc[64 * 72];      // [key in chunk][64 d + 8 pad]  (TDM fills)
    __bf16 Vtc[64 * 72];     // [d][64 keys + 8 pad]
    __bf16 STG[8][16 * 40];  // per-wave prob staging [16 rows][32 + pad]
  };
  __shared__ Smem sm;

  const int tid = threadIdx.x;
  const int wg = blockIdx.x;
  const int qc = wg & 7;
  const int bh = wg >> 3;
  const int h = bh & 15;
  const int b = bh >> 4;

  const int wave = tid >> 5;
  const int lane = tid & 31;
  const int lh = lane >> 4;
  const int l16 = lane & 15;
  const int q0 = qc * 128 + wave * 16;

  const size_t rowbase = (size_t)b * 1024 * 3072;
  const __bf16* kbase = qkv + rowbase + 1024 + h * 64;  // K head base
  const __bf16* vbase = qkv + rowbase + 2048 + h * 64;  // V head base

  // Q fragments (A layout), d-chunks [0,32) and [32,64)
  v16bf aq0, aq1;
  {
    const __bf16* qp = qkv + rowbase + (size_t)(q0 + l16) * 3072 + h * 64;
    v8bf x0 = *(const v8bf*)(qp + lh * 16);
    v8bf x1 = *(const v8bf*)(qp + 8 + lh * 16);
    v8bf x2 = *(const v8bf*)(qp + 32 + lh * 16);
    v8bf x3 = *(const v8bf*)(qp + 40 + lh * 16);
    aq0 = make_v16(x0, x1);
    aq1 = make_v16(x2, x3);
  }

  const float scale = 0.125f;  // 1/sqrt(64)
  const float cshift = 5.0f;

  // -------- pass 1: denominators --------
  float rs[8];
#pragma unroll
  for (int i = 0; i < 8; ++i) rs[i] = 0.f;

  for (int c = 0; c < 16; ++c) {
    __syncthreads();  // previous chunk fully consumed
    if (tid == 0) {
      tdm_load_2d(/*Kc at LDS 0*/ 0u, kbase + (size_t)(c * 64) * 3072, 64, 64,
                  3072, /*32DW*/ 4, /*4DW*/ 3);
      __builtin_amdgcn_s_wait_tensorcnt((short)0);
    }
    __syncthreads();
#pragma unroll
    for (int nt = 0; nt < 4; ++nt) {
      const __bf16* pk = &sm.Kc[(nt * 16 + l16) * 72 + lh * 16];
      v16bf b0 = make_v16(*(const v8bf*)pk, *(const v8bf*)(pk + 8));
      v16bf b1 = make_v16(*(const v8bf*)(pk + 32), *(const v8bf*)(pk + 40));
      v8f cc = {};
      cc = WMMA_BF16(aq0, b0, cc);
      cc = WMMA_BF16(aq1, b1, cc);
#pragma unroll
      for (int i = 0; i < 8; ++i) {
        float t = cc[i] * scale + cshift;
        rs[i] += t * t;
      }
    }
  }
  // reduce across the 16 lanes holding one row's columns (wave32 halves)
#pragma unroll
  for (int m = 8; m >= 1; m >>= 1) {
#pragma unroll
    for (int i = 0; i < 8; ++i) rs[i] += __shfl_xor(rs[i], m, 32);
  }
  float inv[8];
#pragma unroll
  for (int i = 0; i < 8; ++i) inv[i] = 1.0f / (rs[i] + 1e-6f);

  // -------- pass 2: prob + prob@V --------
  v8f o0 = {}, o1 = {}, o2 = {}, o3 = {};
  float* pb = prob + (size_t)(b * 16 + h) * 1024 * 1024;

  for (int c = 0; c < 16; ++c) {
    __syncthreads();
    if (tid == 0)  // kick K chunk DMA, overlap with V transpose below
      tdm_load_2d(0u, kbase + (size_t)(c * 64) * 3072, 64, 64, 3072, 4, 3);
#pragma unroll
    for (int it = 0; it < 2; ++it) {
      int idx = tid + it * 256;
      int r = idx >> 3, c8 = (idx & 7) * 8;
      v8bf vv = *(const v8bf*)(vbase + (size_t)(c * 64 + r) * 3072 + c8);
#pragma unroll
      for (int j = 0; j < 8; ++j) sm.Vtc[(c8 + j) * 72 + r] = vv[j];
    }
    if (tid == 0) __builtin_amdgcn_s_wait_tensorcnt((short)0);
    __syncthreads();

#pragma unroll
    for (int half = 0; half < 2; ++half) {
#pragma unroll
      for (int sub = 0; sub < 2; ++sub) {
        int nt = half * 2 + sub;
        const __bf16* pk = &sm.Kc[(nt * 16 + l16) * 72 + lh * 16];
        v16bf b0 = make_v16(*(const v8bf*)pk, *(const v8bf*)(pk + 8));
        v16bf b1 = make_v16(*(const v8bf*)(pk + 32), *(const v8bf*)(pk + 40));
        v8f cc = {};
        cc = WMMA_BF16(aq0, b0, cc);
        cc = WMMA_BF16(aq1, b1, cc);
#pragma unroll
        for (int i = 0; i < 8; ++i) {
          float t = cc[i] * scale + cshift;
          float p = t * t * inv[i];
          int row = q0 + lh * 8 + i;
          pb[(size_t)row * 1024 + c * 64 + nt * 16 + l16] = p;
          sm.STG[wave][(lh * 8 + i) * 40 + sub * 16 + l16] = (__bf16)p;
        }
      }
      // prob A fragment for these 32 keys (row = lane%16, contiguous K chunks)
      const __bf16* sp = &sm.STG[wave][l16 * 40];
      v16bf pa = make_v16(*(const v8bf*)(sp + lh * 16),
                          *(const v8bf*)(sp + 8 + lh * 16));
      {
        const __bf16* vp = &sm.Vtc[(0 * 16 + l16) * 72 + half * 32 + lh * 16];
        o0 = WMMA_BF16(pa, make_v16(*(const v8bf*)vp, *(const v8bf*)(vp + 8)), o0);
      }
      {
        const __bf16* vp = &sm.Vtc[(1 * 16 + l16) * 72 + half * 32 + lh * 16];
        o1 = WMMA_BF16(pa, make_v16(*(const v8bf*)vp, *(const v8bf*)(vp + 8)), o1);
      }
      {
        const __bf16* vp = &sm.Vtc[(2 * 16 + l16) * 72 + half * 32 + lh * 16];
        o2 = WMMA_BF16(pa, make_v16(*(const v8bf*)vp, *(const v8bf*)(vp + 8)), o2);
      }
      {
        const __bf16* vp = &sm.Vtc[(3 * 16 + l16) * 72 + half * 32 + lh * 16];
        o3 = WMMA_BF16(pa, make_v16(*(const v8bf*)vp, *(const v8bf*)(vp + 8)), o3);
      }
    }
  }

  // epilogue: attn[B*T, 1024] bf16, column base h*64 (merged-head layout)
#pragma unroll
  for (int i = 0; i < 8; ++i) {
    int row = q0 + lh * 8 + i;
    size_t base = (size_t)(b * 1024 + row) * 1024 + h * 64;
    attn[base + 0 * 16 + l16] = (__bf16)o0[i];
    attn[base + 1 * 16 + l16] = (__bf16)o1[i];
    attn[base + 2 * 16 + l16] = (__bf16)o2[i];
    attn[base + 3 * 16 + l16] = (__bf16)o3[i];
  }
}

// ---------------------------------------------------------------------------
extern "C" void kernel_launch(void* const* d_in, const int* in_sizes, int n_in,
                              void* d_out, int out_size, void* d_ws,
                              size_t ws_size, hipStream_t stream) {
  (void)in_sizes; (void)n_in; (void)out_size; (void)ws_size;
  const float* x = (const float*)d_in[0];
  const float* w_qkv = (const float*)d_in[1];
  const float* w_o = (const float*)d_in[2];

  char* ws = (char*)d_ws;
  __bf16* xb    = (__bf16*)(ws);                       //  8 MB: x bf16 [4096,1024]
  __bf16* wqb   = (__bf16*)(ws + ((size_t)8 << 20));   //  6 MB: w_qkv bf16
  __bf16* wob   = (__bf16*)(ws + ((size_t)14 << 20));  //  2 MB: w_o bf16
  __bf16* qkvb  = (__bf16*)(ws + ((size_t)16 << 20));  // 24 MB: qkv bf16 [4096,3072]
  __bf16* attnb = (__bf16*)(ws + ((size_t)40 << 20));  //  8 MB: attn bf16 [4096,1024]

  float* out = (float*)d_out;                   // [4,1024,1024]
  float* prob = out + (size_t)4 * 1024 * 1024;  // [4,16,1024,1024]

  const int n_x = 4096 * 1024;
  const int n_wqkv = 1024 * 3072;
  const int n_wo = 1024 * 1024;
  cvt_f32_to_bf16<<<dim3((n_x / 4 + 255) / 256), 256, 0, stream>>>(x, xb, n_x);
  cvt_f32_to_bf16<<<dim3((n_wqkv / 4 + 255) / 256), 256, 0, stream>>>(w_qkv, wqb, n_wqkv);
  cvt_f32_to_bf16<<<dim3((n_wo / 4 + 255) / 256), 256, 0, stream>>>(w_o, wob, n_wo);

  // qkv = x @ w_qkv  -> bf16 [4096, 3072]
  gemm_bf16_kernel<true><<<dim3(3072 / 64, 4096 / 128), 256, 0, stream>>>(
      xb, wqb, (void*)qkvb, 4096, 3072, 1024);

  // attention: writes prob (f32) and attnb (bf16)
  attn2quad_kernel<<<dim3(512), 256, 0, stream>>>(qkvb, prob, attnb);

  // out = attn @ w_o -> f32 [4096, 1024]
  gemm_bf16_kernel<false><<<dim3(1024 / 64, 4096 / 128), 256, 0, stream>>>(
      attnb, wob, (void*)out, 4096, 1024, 1024);
}
